// MultiHeadAttentionwithRoPE_37357625541121
// MI455X (gfx1250) — matrix operand order repaired and checked by reading, not compiled, not run on hardware
//
#include <hip/hip_runtime.h>

// MultiHeadAttention with RoPE for MI455X (gfx1250, wave32, WMMA).
// B=2, S=2048, DIM=1024, H=16, HD=64.  All GEMMs via v_wmma_f32_16x16x32_bf16.
// Flash attention stages V^T tiles into LDS with double-buffered
// global_load_async_to_lds_b128 (ASYNCcnt), overlapping with QK^T WMMAs.
// Workspace (unsigned short = bf16 bits), 40 MB:
//   [wq|wk|wv|wo] 4 x 1M elems, then q, k, v^T, attn_out 4 x 4M elems.

#define DEV __device__ __forceinline__

typedef __attribute__((ext_vector_type(16))) __bf16 bf16x16;
typedef __attribute__((ext_vector_type(8)))  __bf16 bf16x8;
typedef __attribute__((ext_vector_type(8)))  float  v8f;

union V16 {
  bf16x16 v;
  bf16x8  h[2];
  __bf16  b[16];
  unsigned short u[16];
};

DEV __bf16 f2b(float f) { return (__bf16)f; }          // native v_cvt path
DEV void st_bf(unsigned short* p, float f) { *(__bf16*)p = (__bf16)f; }

DEV v8f wmma_bf(bf16x16 a, bf16x16 b, v8f c) {
  // (neg_a, A, neg_b, B, c_mod, C, reuse_a, reuse_b)
  return __builtin_amdgcn_wmma_f32_16x16x32_bf16(false, a, false, b, (short)0, c,
                                                 false, false);
}

DEV unsigned lds_off(const void* p) {                  // low 32b of generic LDS ptr
  return (unsigned)(unsigned long long)p;
}

// ---------------------------------------------------------------- phase 1
__global__ __launch_bounds__(256) void convert_w_kernel(
    const float4* __restrict__ wq, const float4* __restrict__ wk,
    const float4* __restrict__ wv, const float4* __restrict__ wo,
    unsigned short* __restrict__ dst) {
  unsigned i = blockIdx.x * 256u + threadIdx.x;        // 0 .. 2^20-1 float4s
  unsigned sel = i >> 18;                              // 2^18 float4 per matrix
  const float4* s = (sel == 0) ? wq : (sel == 1) ? wk : (sel == 2) ? wv : wo;
  float4 v = s[i & 0x3FFFFu];
  __bf16* d = (__bf16*)dst + (size_t)i * 4;
  d[0] = (__bf16)v.x; d[1] = (__bf16)v.y; d[2] = (__bf16)v.z; d[3] = (__bf16)v.w;
}

// ---------------------------------------------------------------- phase 2
// C(32x64 per wave) = x @ W^T, RoPE fused for Q/K, V stored transposed.
// grid (32,16,3), block 128 (4 waves -> 128 rows x 64 cols per block).
__global__ __launch_bounds__(128) void qkv_rope_kernel(
    const float* __restrict__ x,
    const unsigned short* __restrict__ wq,
    const unsigned short* __restrict__ wk,
    const unsigned short* __restrict__ wv,
    unsigned short* __restrict__ qb,
    unsigned short* __restrict__ kb,
    unsigned short* __restrict__ vtb) {
  const int lane = threadIdx.x & 31, wave = threadIdx.x >> 5;
  const int lm = lane & 15, lh = lane >> 4;
  const int m0 = blockIdx.x * 128 + wave * 32;         // token-row tile (32 rows)
  const int n0 = blockIdx.y * 64;                      // feature tile
  const int z  = blockIdx.z;                           // 0=Q 1=K 2=V
  const unsigned short* W = (z == 0) ? wq : (z == 1) ? wk : wv;

  v8f acc[2][4] = {};
  for (int kk = 0; kk < 1024; kk += 32) {
    // A tiles (16x32 bf16 each): lane lm = row; K chunks 8*lh and 8*lh+16
    V16 a[2];
#pragma unroll
    for (int r = 0; r < 2; ++r) {
      const float* ar = x + (size_t)(m0 + r * 16 + lm) * 1024 + kk + 8 * lh;
      __builtin_prefetch(ar + 64, 0, 0);               // global_prefetch_b8
#pragma unroll
      for (int e = 0; e < 8; ++e) a[r].b[e]     = f2b(ar[e]);
#pragma unroll
      for (int e = 0; e < 8; ++e) a[r].b[8 + e] = f2b(ar[16 + e]);
    }
    // B tiles (32x16 bf16): lane lm = column n; K chunks 8*lh and 8*lh+16
#pragma unroll
    for (int c = 0; c < 4; ++c) {
      const unsigned short* wr =
          W + (size_t)(n0 + c * 16 + lm) * 1024 + kk + 8 * lh;
      V16 bv; bv.h[0] = *(const bf16x8*)wr; bv.h[1] = *(const bf16x8*)(wr + 16);
#pragma unroll
      for (int r = 0; r < 2; ++r) acc[r][c] = wmma_bf(a[r].v, bv.v, acc[r][c]);
    }
  }

  const float LOG_THETA = 9.210340371976184f;          // ln(10000)
#pragma unroll
  for (int r = 0; r < 2; ++r) {
#pragma unroll
    for (int c = 0; c < 4; ++c) {
      const int col = n0 + c * 16 + lm;                // 0..1023
      if (z < 2) {                                     // RoPE on Q, K
        const int hd = col & 63;
        const float inv_freq =
            __expf(-LOG_THETA * (float)(hd & ~1) * (1.0f / 64.0f));
#pragma unroll
        for (int e = 0; e < 8; ++e) {
          float v = acc[r][c][e];
          float partner = __shfl_xor(v, 1, 32);        // even/odd col pair
          const int row = m0 + r * 16 + e + 8 * lh;
          float ang = (float)(row & 2047) * inv_freq;  // pos = row % S
          float sn, cs; __sincosf(ang, &sn, &cs);
          acc[r][c][e] = (hd & 1) ? (partner * sn + v * cs)
                                  : (v * cs - partner * sn);
        }
      }
#pragma unroll
      for (int e = 0; e < 8; ++e) {
        const int row = m0 + r * 16 + e + 8 * lh;
        if (z == 0)      st_bf(&qb[(size_t)row * 1024 + col], acc[r][c][e]);
        else if (z == 1) st_bf(&kb[(size_t)row * 1024 + col], acc[r][c][e]);
        else {                                         // V transposed (b,h,d,s)
          const int bb = row >> 11, s = row & 2047;
          const int hh = col >> 6,  d = col & 63;
          st_bf(&vtb[(((size_t)(bb * 16 + hh) * 64 + d) << 11) + s],
                acc[r][c][e]);
        }
      }
    }
  }
}

// ---------------------------------------------------------------- phase 3
// Flash attention: one wave per 16-query tile, 32 keys/step, online softmax.
// V^T tiles double-buffered into LDS via global_load_async_to_lds_b128.
// grid (32,16,2), block 128 (4 independent waves; per-wave LDS regions).
DEV void async_v_tile(const unsigned short* vt_base, int kt,
                      unsigned lds_byte_base, int lane) {
  // tile = 64 head-dims x 32 keys (bf16); lane copies rows d=lane, lane+32;
  // each row 64B = 4 x b128 async transfers -> 8 ASYNCcnt ticks per wave.
#pragma unroll
  for (int rr = 0; rr < 2; ++rr) {
    const int d = lane + rr * 32;
    const unsigned short* g = vt_base + ((size_t)d << 11) + kt;
    const unsigned l = lds_byte_base + d * 64;
#pragma unroll
    for (int q = 0; q < 4; ++q) {
      asm volatile("global_load_async_to_lds_b128 %0, %1, off"
                   :: "v"(l + q * 16), "v"(g + q * 8) : "memory");
    }
  }
}

__global__ __launch_bounds__(128) void flash_kernel(
    const unsigned short* __restrict__ qb,
    const unsigned short* __restrict__ kb,
    const unsigned short* __restrict__ vtb,
    unsigned short* __restrict__ attn) {
  __shared__ __align__(16) unsigned short pst[4][16 * 32];      // P per wave
  __shared__ __align__(16) unsigned short vst[4][2][64 * 32];   // V dbl-buf
  const int lane = threadIdx.x & 31, wave = threadIdx.x >> 5;
  const int lm = lane & 15, lh = lane >> 4;
  const int q0 = (blockIdx.x * 4 + wave) * 16;
  const int h = blockIdx.y, b = blockIdx.z;

  // Q as two A tiles (16x32 each, head-dim split)
  bf16x16 aQ[2];
  {
    const unsigned short* qr = qb + (size_t)(b * 2048 + q0 + lm) * 1024 + h * 64;
#pragma unroll
    for (int j = 0; j < 2; ++j) {
      const unsigned short* p = qr + j * 32 + 8 * lh;
      V16 t; t.h[0] = *(const bf16x8*)p; t.h[1] = *(const bf16x8*)(p + 16);
      aQ[j] = t.v;
    }
  }

  v8f o[4] = {};
  float mi[8], li[8];
#pragma unroll
  for (int e = 0; e < 8; ++e) { mi[e] = -__builtin_inff(); li[e] = 0.f; }

  unsigned short* lds = pst[wave];
  const int qmax = q0 + 15;
  const unsigned short* vt_base = vtb + ((size_t)(b * 16 + h) << 17); // *64*2048
  const unsigned vls[2] = { lds_off(vst[wave][0]), lds_off(vst[wave][1]) };

  async_v_tile(vt_base, 0, vls[0], lane);              // prime stage 0
  int parity = 0;

  for (int kt = 0; kt <= qmax; kt += 32) {
    const bool has_next = (kt + 32) <= qmax;
    if (has_next) async_v_tile(vt_base, kt + 32, vls[parity ^ 1], lane);

    float sc[2][8];
#pragma unroll
    for (int t = 0; t < 2; ++t) {                      // two 16-key score tiles
      v8f s = {};
      const int key = kt + t * 16 + lm;                // this lane's key column
      const unsigned short* kr = kb + (size_t)(b * 2048 + key) * 1024 + h * 64;
#pragma unroll
      for (int j = 0; j < 2; ++j) {                    // head-dim K chunks
        const unsigned short* p = kr + j * 32 + 8 * lh;
        V16 tb; tb.h[0] = *(const bf16x8*)p; tb.h[1] = *(const bf16x8*)(p + 16);
        s = wmma_bf(aQ[j], tb.v, s);
      }
#pragma unroll
      for (int e = 0; e < 8; ++e) {
        const int row = q0 + e + 8 * lh;
        sc[t][e] = (key <= row) ? s[e] * 0.125f : -__builtin_inff();
      }
    }
    // online softmax: rows live across 16-lane halves -> shfl_xor reductions
    float al[8];
#pragma unroll
    for (int e = 0; e < 8; ++e) {
      float mx = fmaxf(sc[0][e], sc[1][e]);
#pragma unroll
      for (int off = 8; off >= 1; off >>= 1)
        mx = fmaxf(mx, __shfl_xor(mx, off, 32));
      const float mnew = fmaxf(mi[e], mx);
      al[e] = __expf(mi[e] - mnew);
      const float p0 = __expf(sc[0][e] - mnew);
      const float p1 = __expf(sc[1][e] - mnew);
      sc[0][e] = p0; sc[1][e] = p1;
      float sum = p0 + p1;
#pragma unroll
      for (int off = 8; off >= 1; off >>= 1) sum += __shfl_xor(sum, off, 32);
      li[e] = li[e] * al[e] + sum;
      mi[e] = mnew;
    }
#pragma unroll
    for (int c = 0; c < 4; ++c)
#pragma unroll
      for (int e = 0; e < 8; ++e) o[c][e] *= al[e];

    // D-layout -> A-layout transpose of P through per-wave LDS
#pragma unroll
    for (int t = 0; t < 2; ++t)
#pragma unroll
      for (int e = 0; e < 8; ++e)
        st_bf(&lds[(e + 8 * lh) * 32 + t * 16 + lm], sc[t][e]);
    asm volatile("s_wait_dscnt 0x0" ::: "memory");     // same-wave LDS RAW
    __builtin_amdgcn_wave_barrier();
    bf16x16 aP;
    {
      const unsigned short* p = lds + lm * 32 + 8 * lh;
      V16 t; t.h[0] = *(const bf16x8*)p; t.h[1] = *(const bf16x8*)(p + 16);
      aP = t.v;
    }

    // retire current V tile's 8 async transfers (next tile's 8 stay in flight)
    if (has_next) asm volatile("s_wait_asynccnt 0x8" ::: "memory");
    else          asm volatile("s_wait_asynccnt 0x0" ::: "memory");
    __builtin_amdgcn_wave_barrier();

    // P(16x32) x V(32x64), V from LDS stage [d*32 + k] (bf16)
    const unsigned short* vbuf = vst[wave][parity];
#pragma unroll
    for (int c = 0; c < 4; ++c) {
      const unsigned short* vl = vbuf + (c * 16 + lm) * 32 + 8 * lh;
      V16 tb; tb.h[0] = *(const bf16x8*)vl; tb.h[1] = *(const bf16x8*)(vl + 16);
      o[c] = wmma_bf(aP, tb.v, o[c]);
    }
    parity ^= 1;
  }
#pragma unroll
  for (int e = 0; e < 8; ++e) li[e] = 1.0f / li[e];
#pragma unroll
  for (int c = 0; c < 4; ++c)
#pragma unroll
    for (int e = 0; e < 8; ++e)
      st_bf(&attn[(size_t)(b * 2048 + q0 + e + 8 * lh) * 1024 +
                  h * 64 + c * 16 + lm],
            o[c][e] * li[e]);
}

// ---------------------------------------------------------------- phase 4
// out(f32) = attn(bf16) @ Wo^T.  32x64 per wave.  grid (32,16), block 128.
__global__ __launch_bounds__(128) void oproj_kernel(
    const unsigned short* __restrict__ attn,
    const unsigned short* __restrict__ wob,
    float* __restrict__ out) {
  const int lane = threadIdx.x & 31, wave = threadIdx.x >> 5;
  const int lm = lane & 15, lh = lane >> 4;
  const int m0 = blockIdx.x * 128 + wave * 32;
  const int n0 = blockIdx.y * 64;
  v8f acc[2][4] = {};
  for (int kk = 0; kk < 1024; kk += 32) {
    V16 a[2];
#pragma unroll
    for (int r = 0; r < 2; ++r) {
      const unsigned short* ar =
          attn + (size_t)(m0 + r * 16 + lm) * 1024 + kk + 8 * lh;
      __builtin_prefetch(ar + 64, 0, 0);
      a[r].h[0] = *(const bf16x8*)ar; a[r].h[1] = *(const bf16x8*)(ar + 16);
    }
#pragma unroll
    for (int c = 0; c < 4; ++c) {
      const unsigned short* wr =
          wob + (size_t)(n0 + c * 16 + lm) * 1024 + kk + 8 * lh;
      V16 bv; bv.h[0] = *(const bf16x8*)wr; bv.h[1] = *(const bf16x8*)(wr + 16);
#pragma unroll
      for (int r = 0; r < 2; ++r) acc[r][c] = wmma_bf(a[r].v, bv.v, acc[r][c]);
    }
  }
#pragma unroll
  for (int r = 0; r < 2; ++r)
#pragma unroll
    for (int c = 0; c < 4; ++c)
#pragma unroll
      for (int e = 0; e < 8; ++e)
        out[(size_t)(m0 + r * 16 + e + 8 * lh) * 1024 + n0 + c * 16 + lm] =
            acc[r][c][e];
}

// ---------------------------------------------------------------- launch
extern "C" void kernel_launch(void* const* d_in, const int* in_sizes, int n_in,
                              void* d_out, int out_size, void* d_ws,
                              size_t ws_size, hipStream_t stream) {
  (void)in_sizes; (void)n_in; (void)out_size; (void)ws_size;
  const float* x  = (const float*)d_in[0];
  const float* wq = (const float*)d_in[1];
  const float* wk = (const float*)d_in[2];
  const float* wv = (const float*)d_in[3];
  const float* wo = (const float*)d_in[4];
  float* out = (float*)d_out;

  unsigned short* ws = (unsigned short*)d_ws;
  const size_t MW = 1024u * 1024u;        // weight matrix elems
  const size_t MQ = 4096ull * 1024u;      // activation matrix elems
  unsigned short* wqb  = ws;
  unsigned short* wkb  = wqb + MW;
  unsigned short* wvb  = wkb + MW;
  unsigned short* wob  = wvb + MW;
  unsigned short* qb   = wob + MW;
  unsigned short* kb   = qb + MQ;
  unsigned short* vtb  = kb + MQ;
  unsigned short* attn = vtb + MQ;        // total 40 MB of ws

  convert_w_kernel<<<dim3(4096), dim3(256), 0, stream>>>(
      (const float4*)wq, (const float4*)wk, (const float4*)wv,
      (const float4*)wo, wqb);
  qkv_rope_kernel<<<dim3(32, 16, 3), dim3(128), 0, stream>>>(x, wqb, wkb, wvb,
                                                             qb, kb, vtb);
  flash_kernel<<<dim3(32, 16, 2), dim3(128), 0, stream>>>(qb, kb, vtb, attn);
  oproj_kernel<<<dim3(32, 16, 1), dim3(128), 0, stream>>>(attn, wob, out);
}
